// ColBERT_57904749085358
// MI455X (gfx1250) — matrix operand order repaired and checked by reading, not compiled, not run on hardware
//
#include <hip/hip_runtime.h>
#include <math.h>

// ---------------- problem constants ----------------
#define B_    32
#define N_P   4
#define L_Q   64
#define L_D   1024
#define HID   768
#define DIM   128
#define BN_   (B_ * N_P)      // 128 docs
#define KSTEPS (HID / 32)     // 24

typedef __bf16 bf16_t;
typedef __attribute__((ext_vector_type(16))) __bf16 v16bf;
typedef __attribute__((ext_vector_type(8)))  float  v8f;
typedef int v4i __attribute__((vector_size(16)));

#define AS1 __attribute__((address_space(1)))
#define AS3 __attribute__((address_space(3)))

#if defined(__HIP_DEVICE_COMPILE__) && __has_builtin(__builtin_amdgcn_global_load_async_to_lds_b128)
#define HAVE_ASYNC_LDS 1
#else
#define HAVE_ASYNC_LDS 0
#endif

union BF4 { bf16_t b[4]; uint2 u; };

// ---------------- helpers ----------------
__device__ __forceinline__ void split2(float x, bf16_t& hi, bf16_t& lo) {
    hi = (bf16_t)x;
    lo = (bf16_t)(x - (float)hi);
}

__device__ __forceinline__ v8f wmma_bf16(v16bf a, v16bf b, v8f c) {
    return __builtin_amdgcn_wmma_f32_16x16x32_bf16(
        /*neg_a=*/false, a, /*neg_b=*/false, b,
        /*c_mod=*/(short)0, c, /*reuse_a=*/false, /*reuse_b=*/false);
}

// A-matrix fragment (16x32 bf16, wave32 layout):
// lane l: M = l&15, half = l>>4.  VGPR0-3 hold K = half*8 + 0..7 (contig 16B),
// VGPR4-7 hold K = 16 + half*8 + 0..7 (contig 16B).
__device__ __forceinline__ v16bf load_frag_a(const bf16_t* base, int ldk,
                                             int row, int kbase, int half) {
    const bf16_t* p0 = base + (size_t)row * ldk + kbase + half * 8;
    v16bf r;
    ((int4*)&r)[0] = *(const int4*)(p0);
    ((int4*)&r)[1] = *(const int4*)(p0 + 16);
    return r;
}

// B-matrix fragment (32x16 bf16) sourced from row-major S[N][K] (B[k][n]=S[n][k]):
// lane l: N = l&15, half = l>>4.  VGPR0-3 hold K = half*8 + 0..7, VGPR4-7 hold K = 16 + half*8 + 0..7.
__device__ __forceinline__ v16bf load_frag_b(const bf16_t* base, int ldk,
                                             int col, int kbase, int half) {
    const bf16_t* p0 = base + (size_t)col * ldk + kbase + half * 8;
    v16bf r;
    ((int4*)&r)[0] = *(const int4*)(p0);
    ((int4*)&r)[1] = *(const int4*)(p0 + 16);
    return r;
}

__device__ __forceinline__ v8f zero_v8f() {
    v8f z;
#pragma unroll
    for (int i = 0; i < 8; ++i) z[i] = 0.0f;
    return z;
}

// sign-aware float atomic max (table initialized to -inf)
__device__ __forceinline__ void atomicMaxF(float* addr, float val) {
    if (val != val) return;
    if (val >= 0.0f) atomicMax((int*)addr, __float_as_int(val));
    else             atomicMin((unsigned int*)addr, __float_as_uint(val));
}

// stage one 128x32 bf16 W slice (hi+lo) into LDS; 128 threads, 4 int4 each per array
__device__ __forceinline__ void stage_W(const bf16_t* __restrict__ Whi,
                                        const bf16_t* __restrict__ Wlo,
                                        bf16_t* sW_hi, bf16_t* sW_lo,
                                        int k0, int t) {
#if HAVE_ASYNC_LDS
#pragma unroll
    for (int i = 0; i < 4; ++i) {
        int idx = t + i * 128;              // 0..511 (128 rows x 4 int4)
        int r = idx >> 2, c = idx & 3;
        __builtin_amdgcn_global_load_async_to_lds_b128(
            (AS1 v4i*)(v4i*)&Whi[(size_t)r * HID + k0 + c * 8],
            (AS3 v4i*)(v4i*)&sW_hi[r * 32 + c * 8], 0, 0);
        __builtin_amdgcn_global_load_async_to_lds_b128(
            (AS1 v4i*)(v4i*)&Wlo[(size_t)r * HID + k0 + c * 8],
            (AS3 v4i*)(v4i*)&sW_lo[r * 32 + c * 8], 0, 0);
    }
    asm volatile("s_wait_asynccnt 0x0" ::: "memory");
#else
#pragma unroll
    for (int i = 0; i < 4; ++i) {
        int idx = t + i * 128;
        int r = idx >> 2, c = idx & 3;
        *(int4*)&sW_hi[r * 32 + c * 8] = *(const int4*)&Whi[(size_t)r * HID + k0 + c * 8];
        *(int4*)&sW_lo[r * 32 + c * 8] = *(const int4*)&Wlo[(size_t)r * HID + k0 + c * 8];
    }
#endif
}

// ---------------- kernel 1: split W into bf16 hi/lo (vectorized) ----------------
__global__ __launch_bounds__(256) void k_splitW(const float* __restrict__ W,
                                               bf16_t* __restrict__ Whi,
                                               bf16_t* __restrict__ Wlo) {
    int i = blockIdx.x * 256 + threadIdx.x;    // one float4 per thread
    if (i < DIM * HID / 4) {
        float4 x = ((const float4*)W)[i];
        BF4 h4, l4;
        split2(x.x, h4.b[0], l4.b[0]);
        split2(x.y, h4.b[1], l4.b[1]);
        split2(x.z, h4.b[2], l4.b[2]);
        split2(x.w, h4.b[3], l4.b[3]);
        ((uint2*)Whi)[i] = h4.u;
        ((uint2*)Wlo)[i] = l4.u;
    }
}

// ---------------- kernel 2: init max table to -inf ----------------
__global__ __launch_bounds__(256) void k_init(float* __restrict__ simmax) {
    int i = blockIdx.x * 256 + threadIdx.x;
    if (i < BN_ * L_Q) simmax[i] = -__builtin_inff();
}

// ---------------- kernel 3: queries: project + l2norm -> bf16 hi/lo ----------
// grid = 32 (batch), block = 128 (4 waves, 16 query rows each)
__global__ __launch_bounds__(128) void k_query(const float* __restrict__ Qhid,
                                               const bf16_t* __restrict__ Whi,
                                               const bf16_t* __restrict__ Wlo,
                                               bf16_t* __restrict__ qhi,
                                               bf16_t* __restrict__ qlo) {
    __shared__ __align__(16) bf16_t sA_hi[L_Q * 32];
    __shared__ __align__(16) bf16_t sA_lo[L_Q * 32];
    __shared__ __align__(16) bf16_t sW_hi[DIM * 32];
    __shared__ __align__(16) bf16_t sW_lo[DIM * 32];

    const int b    = blockIdx.x;
    const int t    = threadIdx.x;
    const int wave = t >> 5;
    const int lane = t & 31;
    const int lrow = lane & 15;
    const int half = lane >> 4;

    v8f acc[8];
#pragma unroll
    for (int n = 0; n < 8; ++n) acc[n] = zero_v8f();

    for (int kb = 0; kb < KSTEPS; ++kb) {
        const int k0 = kb * 32;
        // stage 64x32 Q slice (float4 loads, packed bf16x4 stores)
#pragma unroll
        for (int i = 0; i < 4; ++i) {
            int idx = t + i * 128;              // 0..511 (64 rows x 8 float4)
            int r = idx >> 3, c = idx & 7;
            float4 x = *(const float4*)&Qhid[((size_t)b * L_Q + r) * HID + k0 + c * 4];
            BF4 h4, l4;
            split2(x.x, h4.b[0], l4.b[0]);
            split2(x.y, h4.b[1], l4.b[1]);
            split2(x.z, h4.b[2], l4.b[2]);
            split2(x.w, h4.b[3], l4.b[3]);
            *(uint2*)&sA_hi[r * 32 + c * 4] = h4.u;
            *(uint2*)&sA_lo[r * 32 + c * 4] = l4.u;
        }
        stage_W(Whi, Wlo, sW_hi, sW_lo, k0, t);
        __syncthreads();

        v16bf ahi = load_frag_a(sA_hi, 32, wave * 16 + lrow, 0, half);
        v16bf alo = load_frag_a(sA_lo, 32, wave * 16 + lrow, 0, half);
#pragma unroll
        for (int n = 0; n < 8; ++n) {
            v16bf bhi = load_frag_b(sW_hi, 32, n * 16 + lrow, 0, half);
            v16bf blo = load_frag_b(sW_lo, 32, n * 16 + lrow, 0, half);
            acc[n] = wmma_bf16(ahi, bhi, acc[n]);
            acc[n] = wmma_bf16(ahi, blo, acc[n]);
            acc[n] = wmma_bf16(alo, bhi, acc[n]);
        }
        __syncthreads();
    }

    // l2 normalize rows (C-layout: lane holds N=lrow, row M = v + 8*half)
#pragma unroll
    for (int v = 0; v < 8; ++v) {
        float ss = 0.0f;
#pragma unroll
        for (int n = 0; n < 8; ++n) { float x = acc[n][v]; ss += x * x; }
        ss += __shfl_xor(ss, 1, 32);
        ss += __shfl_xor(ss, 2, 32);
        ss += __shfl_xor(ss, 4, 32);
        ss += __shfl_xor(ss, 8, 32);
        float sc = 1.0f / fmaxf(sqrtf(ss), 1e-12f);
        int qrow = b * L_Q + wave * 16 + v + 8 * half;
#pragma unroll
        for (int n = 0; n < 8; ++n) {
            float x = acc[n][v] * sc;
            bf16_t h, l; split2(x, h, l);
            int dimi = n * 16 + lrow;
            qhi[(size_t)qrow * DIM + dimi] = h;
            qlo[(size_t)qrow * DIM + dimi] = l;
        }
    }
}

// ---------------- kernel 4: docs fused project+norm+maxsim ----------------
// grid = (16 token-chunks, 128 docs), block = 128 (4 waves x 16 tokens)
__global__ __launch_bounds__(128) void k_doc(const float* __restrict__ Dhid,
                                             const bf16_t* __restrict__ Whi,
                                             const bf16_t* __restrict__ Wlo,
                                             const bf16_t* __restrict__ qhi,
                                             const bf16_t* __restrict__ qlo,
                                             const unsigned char* __restrict__ dmask,
                                             float* __restrict__ simmax) {
    __shared__ __align__(16) bf16_t sD_hi[64 * 32];
    __shared__ __align__(16) bf16_t sD_lo[64 * 32];
    __shared__ __align__(16) bf16_t sW_hi[DIM * 32];
    __shared__ __align__(16) bf16_t sW_lo[DIM * 32];
    __shared__ __align__(16) bf16_t sP_hi[64 * DIM];
    __shared__ __align__(16) bf16_t sP_lo[64 * DIM];

    const int chunk = blockIdx.x;       // 0..15  (64 tokens each)
    const int bn    = blockIdx.y;       // 0..127
    const int b     = bn >> 2;
    const int t     = threadIdx.x;
    const int wave  = t >> 5;
    const int lane  = t & 31;
    const int lrow  = lane & 15;
    const int half  = lane >> 4;
    const int tok0  = chunk * 64;

    // ---- phase 1: projection P = D_tile @ W^T (bf16x3 compensated) ----
    v8f acc[8];
#pragma unroll
    for (int n = 0; n < 8; ++n) acc[n] = zero_v8f();

    for (int kb = 0; kb < KSTEPS; ++kb) {
        const int k0 = kb * 32;
        // prefetch next D slice while this one converts/computes
        if (kb + 1 < KSTEPS) {
#pragma unroll
            for (int i = 0; i < 4; ++i) {
                int idx = t + i * 128;
                int r = idx >> 3, c = idx & 7;
                __builtin_prefetch(
                    &Dhid[((size_t)bn * L_D + tok0 + r) * HID + k0 + 32 + c * 4], 0, 0);
            }
        }
        // stage 64x32 D slice (float4 loads, packed bf16x4 stores)
#pragma unroll
        for (int i = 0; i < 4; ++i) {
            int idx = t + i * 128;              // 0..511 (64 rows x 8 float4)
            int r = idx >> 3, c = idx & 7;
            float4 x = *(const float4*)&Dhid[((size_t)bn * L_D + tok0 + r) * HID + k0 + c * 4];
            BF4 h4, l4;
            split2(x.x, h4.b[0], l4.b[0]);
            split2(x.y, h4.b[1], l4.b[1]);
            split2(x.z, h4.b[2], l4.b[2]);
            split2(x.w, h4.b[3], l4.b[3]);
            *(uint2*)&sD_hi[r * 32 + c * 4] = h4.u;
            *(uint2*)&sD_lo[r * 32 + c * 4] = l4.u;
        }
        stage_W(Whi, Wlo, sW_hi, sW_lo, k0, t);
        __syncthreads();

        v16bf ahi = load_frag_a(sD_hi, 32, wave * 16 + lrow, 0, half);
        v16bf alo = load_frag_a(sD_lo, 32, wave * 16 + lrow, 0, half);
#pragma unroll
        for (int n = 0; n < 8; ++n) {
            v16bf bhi = load_frag_b(sW_hi, 32, n * 16 + lrow, 0, half);
            v16bf blo = load_frag_b(sW_lo, 32, n * 16 + lrow, 0, half);
            acc[n] = wmma_bf16(ahi, bhi, acc[n]);
            acc[n] = wmma_bf16(ahi, blo, acc[n]);
            acc[n] = wmma_bf16(alo, bhi, acc[n]);
        }
        __syncthreads();
    }

    // ---- phase 2: l2 normalize rows, write P to LDS as bf16 hi/lo ----
#pragma unroll
    for (int v = 0; v < 8; ++v) {
        float ss = 0.0f;
#pragma unroll
        for (int n = 0; n < 8; ++n) { float x = acc[n][v]; ss += x * x; }
        ss += __shfl_xor(ss, 1, 32);
        ss += __shfl_xor(ss, 2, 32);
        ss += __shfl_xor(ss, 4, 32);
        ss += __shfl_xor(ss, 8, 32);
        float sc = 1.0f / fmaxf(sqrtf(ss), 1e-12f);
        int tl = wave * 16 + v + 8 * half;      // token-local row
#pragma unroll
        for (int n = 0; n < 8; ++n) {
            float x = acc[n][v] * sc;
            bf16_t h, l; split2(x, h, l);
            int dimi = n * 16 + lrow;
            sP_hi[tl * DIM + dimi] = h;
            sP_lo[tl * DIM + dimi] = l;
        }
    }
    __syncthreads();

    // ---- phase 3: sim = Q @ P^T, masked max over tokens, atomic update ----
    const bool valid =
        dmask[(size_t)bn * L_D + tok0 + wave * 16 + lrow] != 0;

    v8f s[4];
#pragma unroll
    for (int qt = 0; qt < 4; ++qt) s[qt] = zero_v8f();

#pragma unroll
    for (int kb2 = 0; kb2 < 4; ++kb2) {
        v16bf bph = load_frag_b(sP_hi, DIM, wave * 16 + lrow, kb2 * 32, half);
        v16bf bpl = load_frag_b(sP_lo, DIM, wave * 16 + lrow, kb2 * 32, half);
#pragma unroll
        for (int qt = 0; qt < 4; ++qt) {
            v16bf aqh = load_frag_a(qhi, DIM, b * L_Q + qt * 16 + lrow, kb2 * 32, half);
            v16bf aql = load_frag_a(qlo, DIM, b * L_Q + qt * 16 + lrow, kb2 * 32, half);
            s[qt] = wmma_bf16(aqh, bph, s[qt]);
            s[qt] = wmma_bf16(aqh, bpl, s[qt]);
            s[qt] = wmma_bf16(aql, bph, s[qt]);
        }
    }

#pragma unroll
    for (int qt = 0; qt < 4; ++qt) {
#pragma unroll
        for (int v = 0; v < 8; ++v) {
            float x = valid ? s[qt][v] : -__builtin_inff();
            x = fmaxf(x, __shfl_xor(x, 1, 32));
            x = fmaxf(x, __shfl_xor(x, 2, 32));
            x = fmaxf(x, __shfl_xor(x, 4, 32));
            x = fmaxf(x, __shfl_xor(x, 8, 32));
            if (lrow == 0) {
                int q = qt * 16 + v + 8 * half;
                atomicMaxF(&simmax[bn * L_Q + q], x);
            }
        }
    }
}

// ---------------- kernel 5: sum over queries ----------------
__global__ __launch_bounds__(64) void k_final(const float* __restrict__ simmax,
                                              float* __restrict__ out) {
    __shared__ float sred[2];
    int bn = blockIdx.x;
    int t  = threadIdx.x;
    float v = simmax[bn * L_Q + t];
    v += __shfl_xor(v, 1, 32);
    v += __shfl_xor(v, 2, 32);
    v += __shfl_xor(v, 4, 32);
    v += __shfl_xor(v, 8, 32);
    v += __shfl_xor(v, 16, 32);
    if ((t & 31) == 0) sred[t >> 5] = v;
    __syncthreads();
    if (t == 0) out[bn] = sred[0] + sred[1];
}

// ---------------- launcher ----------------
extern "C" void kernel_launch(void* const* d_in, const int* in_sizes, int n_in,
                              void* d_out, int out_size, void* d_ws, size_t ws_size,
                              hipStream_t stream) {
    (void)in_sizes; (void)n_in; (void)out_size; (void)ws_size;
    const float*         Qhid  = (const float*)d_in[0];
    const float*         Dhid  = (const float*)d_in[1];
    const float*         W     = (const float*)d_in[2];
    const unsigned char* dmask = (const unsigned char*)d_in[3];  // bool8
    float* out = (float*)d_out;

    char* ws = (char*)d_ws;
    bf16_t* Whi    = (bf16_t*)(ws);                         // 196608 B
    bf16_t* Wlo    = (bf16_t*)(ws + 196608);                // 196608 B
    bf16_t* qhi    = (bf16_t*)(ws + 393216);                // 524288 B
    bf16_t* qlo    = (bf16_t*)(ws + 917504);                // 524288 B
    float*  simmax = (float*) (ws + 1441792);               //  32768 B

    k_splitW<<<(DIM * HID / 4 + 255) / 256, 256, 0, stream>>>(W, Whi, Wlo);
    k_init  <<<(BN_ * L_Q + 255) / 256, 256, 0, stream>>>(simmax);
    k_query <<<B_, 128, 0, stream>>>(Qhid, Whi, Wlo, qhi, qlo);
    k_doc   <<<dim3(L_D / 64, BN_), 128, 0, stream>>>(Dhid, Whi, Wlo, qhi, qlo,
                                                      dmask, simmax);
    k_final <<<BN_, 64, 0, stream>>>(simmax, out);
}
